// DocVec_26156350832816
// MI455X (gfx1250) — matrix operand order repaired and checked by reading, not compiled, not run on hardware
//
#include <hip/hip_runtime.h>
#include <hip/hip_bf16.h>

#define T_LEN 8192
#define EDIM  256
#define HDIM  256
#define G4H   1024
#define FLEN  5
#define FN    128
#define NC    20
#define KCONV (FLEN*HDIM)        // 1280
#define NPOS  (T_LEN - FLEN + 1) // 8188

typedef __attribute__((ext_vector_type(16))) __bf16 v16bf;
typedef __attribute__((ext_vector_type(8)))  float  v8f;

typedef unsigned short u16;
typedef unsigned int   u32;

// ---------- ordered-float <-> uint for deterministic atomic max ----------
__device__ __forceinline__ u32 f2ord(float f) {
  u32 u = __float_as_uint(f);
  return (u & 0x80000000u) ? ~u : (u | 0x80000000u);
}
__device__ __forceinline__ float ord2f(u32 u) {
  u32 b = (u & 0x80000000u) ? (u & 0x7fffffffu) : ~u;
  return __uint_as_float(b);
}

// ---------- WMMA tile loaders (contiguous-k layouts, 16-bit A / B) ----------
// A (MxK=16x32): lanes 0-15 hold M=lane, K = kb+0..15;
//                lanes 16-31 hold M=lane-16, K = kb+16..31.
__device__ __forceinline__ v16bf load_tileA(const u16* __restrict__ base, int lda,
                                            int mb, int kb, int lane) {
  int l  = lane & 15;
  int hi = (lane >> 4) & 1;
  const u16* p = base + (size_t)(mb + l) * lda + kb + hi * 16;
  union { v16bf v; uint4 q[2]; } u;
  u.q[0] = *(const uint4*)(p);
  u.q[1] = *(const uint4*)(p + 8);
  return u.v;
}
// B (KxN=32x16), weights stored N-major (row n = 16-bit K-contiguous):
// lanes 0-15 hold N=lane, K = kb+0..15; lanes 16-31 hold N=lane-16, K = kb+16..31.
__device__ __forceinline__ v16bf load_tileB(const u16* __restrict__ base, int ldb,
                                            int nb, int kb, int lane) {
  int l  = lane & 15;
  int hi = (lane >> 4) & 1;
  const u16* p = base + (size_t)(nb + l) * ldb + kb + hi * 16;
  union { v16bf v; uint4 q[2]; } u;
  u.q[0] = *(const uint4*)(p);
  u.q[1] = *(const uint4*)(p + 8);
  return u.v;
}

// ---------------------------- prep kernels ----------------------------
__global__ void k_embed_relu(const int* __restrict__ docs,
                             const float* __restrict__ embed_W,
                             __hip_bfloat16* __restrict__ xbf) {
  int idx = blockIdx.x * blockDim.x + threadIdx.x;
  if (idx >= T_LEN * EDIM) return;
  int t = idx >> 8, e = idx & 255;
  float v = embed_W[(size_t)docs[t] * EDIM + e];
  v = v > 0.f ? v : 0.f;
  xbf[idx] = __float2bfloat16(v);
}

__global__ void k_cvt_bf16(const float* __restrict__ in,
                           __hip_bfloat16* __restrict__ out, int n) {
  int i = blockIdx.x * blockDim.x + threadIdx.x;
  if (i < n) out[i] = __float2bfloat16(in[i]);
}

// repack w_hh (1024x256 row-major) into w_p[k4][j][4] so each thread of the
// recurrent workgroup streams its 4 K-values per iter as one coalesced b128.
__global__ void k_pack_whh(const float* __restrict__ whh, float* __restrict__ wp) {
  int o = blockIdx.x * blockDim.x + threadIdx.x;
  if (o >= HDIM * G4H) return;
  int k4 = o >> 12;      // / 4096
  int r  = o & 4095;
  int j  = r >> 2;       // gate output index 0..1023
  int jj = r & 3;        // k sub-index
  wp[o] = whh[j * HDIM + k4 * 4 + jj];
}

__global__ void k_init(__hip_bfloat16* __restrict__ hs_pad, u32* __restrict__ pooled) {
  int i = blockIdx.x * blockDim.x + threadIdx.x;
  if (i < 16 * HDIM)            hs_pad[i] = __float2bfloat16(0.f);
  else if (i < 16 * HDIM + FN)  pooled[i - 16 * HDIM] = 0u;   // ordered-min == -inf
}

// --------------- GEMM 1: xproj = relu(embed) @ w_ih^T + (b_ih+b_hh) ----------
// M=8192, N=1024, K=256. One wave per 16x16 tile.
__global__ void k_gemm_xproj(const u16* __restrict__ xbf,     // (T, E) bf16
                             const u16* __restrict__ wbf,     // (4H, E) bf16, N-major
                             const float* __restrict__ b_ih,
                             const float* __restrict__ b_hh,
                             float* __restrict__ xproj) {     // (T, 4H) f32
  int tile = blockIdx.x;            // (T/16)*(4H/16) = 512*64
  int nt = tile & 63, mt = tile >> 6;
  int lane = threadIdx.x;
  int mb = mt * 16, nb = nt * 16;
  v8f acc = {};
#pragma unroll
  for (int kb = 0; kb < EDIM; kb += 32) {
    v16bf a = load_tileA(xbf, EDIM, mb, kb, lane);
    v16bf b = load_tileB(wbf, EDIM, nb, kb, lane);
    acc = __builtin_amdgcn_wmma_f32_16x16x32_bf16(false, a, false, b,
                                                  (short)0, acc, false, false);
  }
  int l = lane & 15, hi = (lane >> 4) & 1;
  int n = nb + l;
  float bias = b_ih[n] + b_hh[n];
#pragma unroll
  for (int i = 0; i < 8; ++i) {
    int m = mb + i + hi * 8;
    xproj[(size_t)m * G4H + n] = acc[i] + bias;
  }
}

// ------------------- recurrent LSTM: 1 workgroup, 1024 threads ---------------
__global__ void __launch_bounds__(1024)
k_lstm(const float* __restrict__ xproj,      // (T, 4H)
       const float* __restrict__ wp,         // packed w_hh
       __hip_bfloat16* __restrict__ hs_bf) { // (T+16, H) bf16
  __shared__ __align__(16) float h_sh[HDIM];
  __shared__ float g_sh[G4H];
  const int tid = threadIdx.x;
  float c_state = 0.f;
  if (tid < HDIM) h_sh[tid] = 0.f;
  __syncthreads();

  const float4* wq = (const float4*)wp;      // [k4*1024 + tid]
  const float4* h4 = (const float4*)h_sh;

  for (int t = 0; t < T_LEN; ++t) {
    float acc = xproj[(size_t)t * G4H + tid];
    if (t + 1 < T_LEN)
      __builtin_prefetch(&xproj[(size_t)(t + 1) * G4H + tid], 0, 0);
#pragma unroll 16
    for (int k4 = 0; k4 < HDIM / 4; ++k4) {
      float4 w = wq[k4 * G4H + tid];
      float4 h = h4[k4];                      // LDS broadcast
      acc = fmaf(w.x, h.x, acc);
      acc = fmaf(w.y, h.y, acc);
      acc = fmaf(w.z, h.z, acc);
      acc = fmaf(w.w, h.w, acc);
    }
    g_sh[tid] = acc;
    __syncthreads();
    if (tid < HDIM) {
      float gi = 1.f / (1.f + __expf(-g_sh[tid]));
      float gf = 1.f / (1.f + __expf(-g_sh[tid + HDIM]));
      float gg = tanhf(g_sh[tid + 2 * HDIM]);
      float go = 1.f / (1.f + __expf(-g_sh[tid + 3 * HDIM]));
      c_state = gf * c_state + gi * gg;
      float hn = go * tanhf(c_state);
      h_sh[tid] = hn;
      hs_bf[(size_t)t * HDIM + tid] = __float2bfloat16(hn);
    }
    __syncthreads();
  }
}

// --------- GEMM 2 (conv as GEMM) + fused time-max into pooled[FN] -----------
// A[t,k] = hs[t*256 + k] (windows are contiguous, lda = 256, K = 1280)
// B = conv_w reshaped (FN, 1280), N-major. M tiles padded to 8192; mask t>=8188.
__global__ void k_conv_max(const u16* __restrict__ hsbf,
                           const u16* __restrict__ cwbf,
                           const float* __restrict__ conv_b,
                           u32* __restrict__ pooled) {
  int tile = blockIdx.x;            // (8192/16)*(128/16) = 512*8
  int nt = tile & 7, mt = tile >> 3;
  int lane = threadIdx.x;
  int mb = mt * 16, nb = nt * 16;
  v8f acc = {};
#pragma unroll 8
  for (int kb = 0; kb < KCONV; kb += 32) {
    v16bf a = load_tileA(hsbf, HDIM, mb, kb, lane);
    v16bf b = load_tileB(cwbf, KCONV, nb, kb, lane);
    acc = __builtin_amdgcn_wmma_f32_16x16x32_bf16(false, a, false, b,
                                                  (short)0, acc, false, false);
  }
  int l = lane & 15, hi = (lane >> 4) & 1;
  int n = nb + l;
  float cb = conv_b[n];
  float mx = -3.4e38f;
#pragma unroll
  for (int i = 0; i < 8; ++i) {
    int m = mb + i + hi * 8;
    float v = acc[i] + cb;
    if (m < NPOS) mx = fmaxf(mx, v);
  }
  atomicMax(&pooled[n], f2ord(mx));
}

// ------------------------------- finalize -----------------------------------
__global__ void k_final(const u32* __restrict__ pooled,
                        const float* __restrict__ logits_w,
                        const float* __restrict__ logits_b,
                        const int* __restrict__ labels,
                        float* __restrict__ out) {
  __shared__ float p_sh[FN];
  __shared__ float l_sh[NC];
  int tid = threadIdx.x;
  p_sh[tid] = ord2f(pooled[tid]);
  __syncthreads();
  if (tid < NC) {
    float s = logits_b[tid];
#pragma unroll 16
    for (int f = 0; f < FN; ++f) s = fmaf(p_sh[f], logits_w[tid * FN + f], s);
    l_sh[tid] = s;
  }
  __syncthreads();
  if (tid == 0) {
    float mx = l_sh[0]; int am = 0;
    for (int c = 1; c < NC; ++c) if (l_sh[c] > mx) { mx = l_sh[c]; am = c; }
    float se = 0.f;
    for (int c = 0; c < NC; ++c) se += __expf(l_sh[c] - mx);
    float lse = mx + __logf(se);
    int lab = labels[0];
    out[0] = -(l_sh[lab] - lse);
    out[1] = (am == lab) ? 1.f : 0.f;
  }
}

// ------------------------------- launcher -----------------------------------
extern "C" void kernel_launch(void* const* d_in, const int* in_sizes, int n_in,
                              void* d_out, int out_size, void* d_ws, size_t ws_size,
                              hipStream_t stream) {
  const int*   docs     = (const int*)  d_in[0];
  const int*   labels   = (const int*)  d_in[2];
  const float* embed_W  = (const float*)d_in[4];
  const float* w_ih     = (const float*)d_in[5];
  const float* w_hh     = (const float*)d_in[6];
  const float* b_ih     = (const float*)d_in[7];
  const float* b_hh     = (const float*)d_in[8];
  const float* conv_w   = (const float*)d_in[9];
  const float* conv_b   = (const float*)d_in[10];
  const float* logits_w = (const float*)d_in[11];
  const float* logits_b = (const float*)d_in[12];
  float* out = (float*)d_out;

  // workspace carve-up (256B aligned)
  char* ws = (char*)d_ws;
  size_t off = 0;
  auto carve = [&](size_t bytes) -> char* {
    char* p = ws + off;
    off = (off + bytes + 255) & ~(size_t)255;
    return p;
  };
  __hip_bfloat16* x_bf    = (__hip_bfloat16*)carve((size_t)T_LEN * EDIM * 2);
  __hip_bfloat16* wih_bf  = (__hip_bfloat16*)carve((size_t)G4H * EDIM * 2);
  __hip_bfloat16* cw_bf   = (__hip_bfloat16*)carve((size_t)FN * KCONV * 2);
  float*          whh_p   = (float*)carve((size_t)HDIM * G4H * 4);
  float*          xproj   = (float*)carve((size_t)T_LEN * G4H * 4);
  __hip_bfloat16* hs_bf   = (__hip_bfloat16*)carve((size_t)(T_LEN + 16) * HDIM * 2);
  u32*            pooled  = (u32*)carve((size_t)FN * 4);
  (void)ws_size; (void)in_sizes; (void)n_in; (void)out_size;

  // prep
  k_embed_relu<<<(T_LEN * EDIM + 255) / 256, 256, 0, stream>>>(docs, embed_W, x_bf);
  k_cvt_bf16<<<(G4H * EDIM + 255) / 256, 256, 0, stream>>>(w_ih, wih_bf, G4H * EDIM);
  k_cvt_bf16<<<(FN * KCONV + 255) / 256, 256, 0, stream>>>(conv_w, cw_bf, FN * KCONV);
  k_pack_whh<<<(HDIM * G4H + 255) / 256, 256, 0, stream>>>(w_hh, whh_p);
  k_init<<<(16 * HDIM + FN + 255) / 256, 256, 0, stream>>>(hs_bf + (size_t)T_LEN * HDIM, pooled);

  // input projection GEMM (WMMA bf16): 512 x 64 tiles, one wave each
  k_gemm_xproj<<<(T_LEN / 16) * (G4H / 16), 32, 0, stream>>>(
      (const u16*)x_bf, (const u16*)wih_bf, b_ih, b_hh, xproj);

  // sequential recurrence: single persistent workgroup
  k_lstm<<<1, 1024, 0, stream>>>(xproj, whh_p, hs_bf);

  // conv-as-GEMM (WMMA bf16) + fused max pool
  k_conv_max<<<(T_LEN / 16) * (FN / 16), 32, 0, stream>>>(
      (const u16*)hs_bf, (const u16*)cw_bf, conv_b, pooled);

  // logits + log-softmax + loss/acc
  k_final<<<1, FN, 0, stream>>>(pooled, logits_w, logits_b, labels, out);
}
